// SelfAttention_52922587022074
// MI455X (gfx1250) — matrix operand order repaired and checked by reading, not compiled, not run on hardware
//
#include <hip/hip_runtime.h>
#include <hip/hip_bf16.h>

typedef _Float16 v16h __attribute__((ext_vector_type(16)));
typedef _Float16 v8h  __attribute__((ext_vector_type(8)));
typedef float    v8f  __attribute__((ext_vector_type(8)));

#define D_    1024
#define S_    2048
#define B_    2
#define H_    16
#define HD_   64
#define MTOT  (B_ * S_)   // 4096

// -------------------------------------------------------------------------
// fp32 -> fp16 convert (plain)
__global__ __launch_bounds__(256) void cvt_f16(const float* __restrict__ in,
                                               _Float16* __restrict__ out, int n) {
  int i = blockIdx.x * blockDim.x + threadIdx.x;
  if (i < n) out[i] = (_Float16)in[i];
}

// fp32 [1024][1024] -> fp16 transposed [n][k]
__global__ __launch_bounds__(256) void cvt_f16_T(const float* __restrict__ in,
                                                 _Float16* __restrict__ out) {
  int i = blockIdx.x * blockDim.x + threadIdx.x;   // 1M elements
  int k = i >> 10, n = i & 1023;
  out[(size_t)n * D_ + k] = (_Float16)in[i];
}

// -------------------------------------------------------------------------
// WMMA GEMM: A [M=4096][1024] f16 row-major, Bt [N=1024][1024] f16 (W^T).
// Wave tile 32x64 (2 A-frags x 4 B-frags, 8 WMMAs / 32-k-step), 4 waves =
// 128x64 block tile. Explicit ping-pong double buffering (no rotation copies).
// mode 0: out f16 [B,H,S,HD]   (Q, K)
// mode 1: out f16 [B,H,HD,S]   (V, transposed for PV WMMA)
// mode 2: out f32 [M][N] = acc + bias + resid   (output projection + residual)
__global__ __launch_bounds__(128)
void gemm_wmma(const _Float16* __restrict__ A, const _Float16* __restrict__ Bt,
               const float* __restrict__ bias, const float* __restrict__ resid,
               void* __restrict__ out, int mode) {
  const int lane = threadIdx.x & 31;
  const int wave = threadIdx.x >> 5;
  const int hs   = lane >> 4;       // half-wave select
  const int l15  = lane & 15;
  const int m0 = blockIdx.y * 128 + wave * 32;
  const int n0 = blockIdx.x * 64;

  v8f acc[2][4] = {{v8f{}, v8f{}, v8f{}, v8f{}}, {v8f{}, v8f{}, v8f{}, v8f{}}};

  const _Float16* arow[2] = {A + (size_t)(m0 + l15) * D_,
                             A + (size_t)(m0 + 16 + l15) * D_};
  const _Float16* brow[4];
#pragma unroll
  for (int t = 0; t < 4; ++t)
    brow[t] = Bt + (size_t)(n0 + t * 16 + l15) * D_ + hs * 16;

  v16h aF[2][2], bF[2][4];   // [buffer][fragment]

  auto loadBuf = [&](int buf, int k0) {
#pragma unroll
    for (int r = 0; r < 2; ++r) {
      v8h lo = *(const v8h*)(arow[r] + k0 + hs * 16);
      v8h hi = *(const v8h*)(arow[r] + k0 + hs * 16 + 8);
      aF[buf][r] = __builtin_shufflevector(lo, hi, 0,1,2,3,4,5,6,7,8,9,10,11,12,13,14,15);
    }
#pragma unroll
    for (int t = 0; t < 4; ++t) bF[buf][t] = *(const v16h*)(brow[t] + k0);
  };
  auto mmaBuf = [&](int buf) {
#pragma unroll
    for (int t = 0; t < 4; ++t) {
      acc[0][t] = __builtin_amdgcn_wmma_f32_16x16x32_f16(false, aF[buf][0], false, bF[buf][t],
                                                         (short)0, acc[0][t], false, false);
      acc[1][t] = __builtin_amdgcn_wmma_f32_16x16x32_f16(false, aF[buf][1], false, bF[buf][t],
                                                         (short)0, acc[1][t], false, false);
    }
  };

  loadBuf(0, 0);
  for (int k0 = 0; k0 < D_; k0 += 64) {
    loadBuf(1, k0 + 32);                 // prefetch odd step
    __builtin_prefetch(arow[0] + k0 + 256, 0, 3);
    mmaBuf(0);                           // consume even step
    loadBuf(0, (k0 + 64) & (D_ - 1));    // prefetch next even (wraps harmlessly)
    mmaBuf(1);                           // consume odd step
  }

#pragma unroll
  for (int r = 0; r < 2; ++r)
#pragma unroll
    for (int t = 0; t < 4; ++t) {
      const int n = n0 + t * 16 + l15;
      const float bv = bias[n];
#pragma unroll
      for (int j = 0; j < 8; ++j) {
        const int m = m0 + r * 16 + j + 8 * hs;   // C/D layout row
        float v = acc[r][t][j] + bv;
        if (mode == 2) {
          float* o = (float*)out;
          o[(size_t)m * D_ + n] = v + resid[(size_t)m * D_ + n];
        } else {
          const int b = m >> 11, s = m & (S_ - 1);
          const int h = n >> 6, hd = n & (HD_ - 1);
          _Float16* o = (_Float16*)out;
          if (mode == 0)
            o[(((size_t)(b * H_ + h) * S_) + s) * HD_ + hd] = (_Float16)v;
          else
            o[(((size_t)(b * H_ + h) * HD_) + hd) * S_ + s] = (_Float16)v;
        }
      }
    }
}

// -------------------------------------------------------------------------
// Flash attention: one wave owns a 16-query tile; streams keys in 32-chunks.
// Q,K: [B,H,S,64] f16.  Vt: [B,H,64,S] f16.  ctx out: [B,S,1024] f16.
// V loads issue before the softmax block; K fragments are double-buffered so
// softmax + PV WMMAs hide the next tile's K latency.
__global__ __launch_bounds__(128)
void flash_attn(const _Float16* __restrict__ Qh, const _Float16* __restrict__ Kh,
                const _Float16* __restrict__ Vt, _Float16* __restrict__ ctxh) {
  __shared__ _Float16 Plds[4][16][32];          // per-wave P relayout buffer
  const int lane = threadIdx.x & 31;
  const int wave = threadIdx.x >> 5;
  const int hs   = lane >> 4;
  const int l15  = lane & 15;

  const int T  = blockIdx.x * 4 + wave;         // 0..4095 query tiles
  const int bh = T >> 7;                        // 128 tiles per (b,h)
  const int q0 = (T & 127) << 4;

  // load Q as two 16x32 A-fragments (head-dim chunks)
  const _Float16* qrow = Qh + ((size_t)bh * S_ + q0 + l15) * HD_;
  v16h aq[2];
#pragma unroll
  for (int c = 0; c < 2; ++c) {
    v8h lo = *(const v8h*)(qrow + c * 32 + hs * 16);
    v8h hi = *(const v8h*)(qrow + c * 32 + hs * 16 + 8);
    aq[c] = __builtin_shufflevector(lo, hi, 0,1,2,3,4,5,6,7,8,9,10,11,12,13,14,15);
  }

  float rowmax[8], rowsum[8];
#pragma unroll
  for (int j = 0; j < 8; ++j) { rowmax[j] = -3.0e38f; rowsum[j] = 0.f; }
  v8f acc[4] = {v8f{}, v8f{}, v8f{}, v8f{}};
  const float scale = 0.125f;                   // 1/sqrt(64)

  const _Float16* kbase = Kh + (size_t)bh * S_ * HD_;
  const _Float16* vbase = Vt + ((size_t)bh * HD_ + l15) * S_ + hs * 16;

  v16h kf[2][4];   // double-buffered K fragments [buf][g*2+c]
  auto loadK = [&](int buf, int kt) {
#pragma unroll
    for (int g = 0; g < 2; ++g)
#pragma unroll
      for (int c = 0; c < 2; ++c)
        kf[buf][g * 2 + c] = *(const v16h*)(kbase + (size_t)(kt + g * 16 + l15) * HD_
                                            + c * 32 + hs * 16);
  };
  loadK(0, 0);

#pragma unroll 2
  for (int kt = 0; kt < S_; kt += 32) {
    const int cur = (kt >> 5) & 1;     // compile-time constant under unroll 2

    // issue V-tile loads early; consumed only after the softmax block
    v16h bv[4];
#pragma unroll
    for (int t = 0; t < 4; ++t)
      bv[t] = *(const v16h*)(vbase + (size_t)(t * 16) * S_ + kt);

    // scores: two 16x16 tiles (keys kt..+15, kt+16..+31), K=64 in 2 WMMAs each
    v8f s[2] = {v8f{}, v8f{}};
#pragma unroll
    for (int g = 0; g < 2; ++g)
#pragma unroll
      for (int c = 0; c < 2; ++c)
        s[g] = __builtin_amdgcn_wmma_f32_16x16x32_f16(false, aq[c], false, kf[cur][g * 2 + c],
                                                      (short)0, s[g], false, false);

    // preload next tile's K; softmax + PV below hide the latency
    loadK(cur ^ 1, (kt + 32) & (S_ - 1));

    // online softmax per row (row m=j+8*hs lives across the 16 lanes of a half)
    float p0[8], p1[8];
#pragma unroll
    for (int j = 0; j < 8; ++j) {
      float a0 = s[0][j] * scale, a1 = s[1][j] * scale;
      float tm = fmaxf(a0, a1);
#pragma unroll
      for (int off = 1; off < 16; off <<= 1)
        tm = fmaxf(tm, __shfl_xor(tm, off, 32));
      float nm = fmaxf(rowmax[j], tm);
      float corr = __expf(rowmax[j] - nm);
      rowmax[j] = nm;
      float e0 = __expf(a0 - nm), e1 = __expf(a1 - nm);
      float ls = e0 + e1;
#pragma unroll
      for (int off = 1; off < 16; off <<= 1)
        ls += __shfl_xor(ls, off, 32);
      rowsum[j] = rowsum[j] * corr + ls;
      p0[j] = e0; p1[j] = e1;
#pragma unroll
      for (int t = 0; t < 4; ++t) acc[t][j] *= corr;   // rescale running ctx
    }

    // relayout P (16x32) from C/D layout to A-fragment layout via LDS
#pragma unroll
    for (int j = 0; j < 8; ++j) {
      Plds[wave][j + 8 * hs][l15]      = (_Float16)p0[j];
      Plds[wave][j + 8 * hs][16 + l15] = (_Float16)p1[j];
    }
    asm volatile("s_wait_dscnt 0" ::: "memory");       // DS is in-order per wave
    v8h plo = *(const v8h*)&Plds[wave][l15][hs * 16];
    v8h phi = *(const v8h*)&Plds[wave][l15][hs * 16 + 8];
    v16h pa = __builtin_shufflevector(plo, phi, 0,1,2,3,4,5,6,7,8,9,10,11,12,13,14,15);

    // ctx += P (16x32) * V (32x64), 4 WMMAs over dim chunks
#pragma unroll
    for (int t = 0; t < 4; ++t)
      acc[t] = __builtin_amdgcn_wmma_f32_16x16x32_f16(false, pa, false, bv[t],
                                                      (short)0, acc[t], false, false);
  }

  // normalize and store ctx as [B,S,1024] (merge heads)
  const int b = bh >> 4, h = bh & 15;
#pragma unroll
  for (int t = 0; t < 4; ++t)
#pragma unroll
    for (int j = 0; j < 8; ++j) {
      const int q = q0 + j + 8 * hs;
      const int dim = t * 16 + l15;
      float v = acc[t][j] / rowsum[j];
      ctxh[((size_t)(b * S_) + q) * D_ + h * HD_ + dim] = (_Float16)v;
    }
}

// -------------------------------------------------------------------------
// Row LayerNorm: one block per row of y [4096][1024] fp32
__global__ __launch_bounds__(256)
void layernorm(const float* __restrict__ y, const float* __restrict__ gamma,
               const float* __restrict__ beta, float* __restrict__ out) {
  __shared__ float red[256];
  const int m = blockIdx.x, t = threadIdx.x;
  const float* row = y + (size_t)m * D_;
  float v[4]; float s = 0.f;
#pragma unroll
  for (int i = 0; i < 4; ++i) { v[i] = row[t * 4 + i]; s += v[i]; }
  red[t] = s; __syncthreads();
  for (int off = 128; off > 0; off >>= 1) {
    if (t < off) red[t] += red[t + off];
    __syncthreads();
  }
  const float mu = red[0] * (1.f / D_);
  __syncthreads();
  float sq = 0.f;
#pragma unroll
  for (int i = 0; i < 4; ++i) { float d = v[i] - mu; sq += d * d; }
  red[t] = sq; __syncthreads();
  for (int off = 128; off > 0; off >>= 1) {
    if (t < off) red[t] += red[t + off];
    __syncthreads();
  }
  const float rstd = rsqrtf(red[0] * (1.f / D_) + 1e-5f);
#pragma unroll
  for (int i = 0; i < 4; ++i) {
    int c = t * 4 + i;
    out[(size_t)m * D_ + c] = (v[i] - mu) * rstd * gamma[c] + beta[c];
  }
}

// -------------------------------------------------------------------------
extern "C" void kernel_launch(void* const* d_in, const int* in_sizes, int n_in,
                              void* d_out, int out_size, void* d_ws, size_t ws_size,
                              hipStream_t stream) {
  const float* x     = (const float*)d_in[0];
  const float* Wq    = (const float*)d_in[1];
  const float* bq    = (const float*)d_in[2];
  const float* Wk    = (const float*)d_in[3];
  const float* bk    = (const float*)d_in[4];
  const float* Wv    = (const float*)d_in[5];
  const float* bv    = (const float*)d_in[6];
  const float* Wo    = (const float*)d_in[7];
  const float* bo    = (const float*)d_in[8];
  const float* gamma = (const float*)d_in[9];
  const float* beta  = (const float*)d_in[10];

  char* ws = (char*)d_ws;
  _Float16* xh   = (_Float16*)(ws);                       //  8 MiB
  _Float16* Wqt  = (_Float16*)(ws + ( 8ull << 20));       //  2 MiB
  _Float16* Wkt  = (_Float16*)(ws + (10ull << 20));       //  2 MiB
  _Float16* Wvt  = (_Float16*)(ws + (12ull << 20));       //  2 MiB
  _Float16* Wot  = (_Float16*)(ws + (14ull << 20));       //  2 MiB
  _Float16* Qh   = (_Float16*)(ws + (16ull << 20));       //  8 MiB
  _Float16* Kh   = (_Float16*)(ws + (24ull << 20));       //  8 MiB
  _Float16* Vt   = (_Float16*)(ws + (32ull << 20));       //  8 MiB
  _Float16* ctxh = (_Float16*)(ws + (40ull << 20));       //  8 MiB
  float*    y    = (float*)   (ws + (48ull << 20));       // 16 MiB

  const int nx = MTOT * D_;                               // 4194304
  cvt_f16<<<(nx + 255) / 256, 256, 0, stream>>>(x, xh, nx);
  cvt_f16_T<<<4096, 256, 0, stream>>>(Wq, Wqt);
  cvt_f16_T<<<4096, 256, 0, stream>>>(Wk, Wkt);
  cvt_f16_T<<<4096, 256, 0, stream>>>(Wv, Wvt);
  cvt_f16_T<<<4096, 256, 0, stream>>>(Wo, Wot);

  dim3 gg(D_ / 64, MTOT / 128);                           // 16 x 32
  gemm_wmma<<<gg, 128, 0, stream>>>(xh, Wqt, bq, nullptr, Qh, 0);
  gemm_wmma<<<gg, 128, 0, stream>>>(xh, Wkt, bk, nullptr, Kh, 0);
  gemm_wmma<<<gg, 128, 0, stream>>>(xh, Wvt, bv, nullptr, Vt, 1);

  flash_attn<<<1024, 128, 0, stream>>>(Qh, Kh, Vt, ctxh);

  gemm_wmma<<<gg, 128, 0, stream>>>(ctxh, Wot, bo, x, y, 2);
  layernorm<<<MTOT, 256, 0, stream>>>(y, gamma, beta, (float*)d_out);
}
